// Model_83184926589262
// MI455X (gfx1250) — matrix operand order repaired and checked by reading, not compiled, not run on hardware
//
#include <hip/hip_runtime.h>
#include <hip/hip_bf16.h>
#include <stdint.h>

// ---------------------------------------------------------------------------
// MeshGraphNets forward on MI455X (gfx1250): bf16 WMMA GEMMs + fused epilogues
// X tiles staged with CDNA5 async global->LDS (ASYNCcnt); W tiles transposed
// into LDS with packed b32 stores.
// ---------------------------------------------------------------------------

typedef __attribute__((ext_vector_type(16))) __bf16 v16bf;
typedef __attribute__((ext_vector_type(8)))  float  v8f;

#define N_NODES_C 30000
#define N_EDGES_C 180000
#define LATENT_C  128
#define STEPS_C   15
#define LN_EPS_C  1e-5f

__device__ __forceinline__ unsigned short f2bf(float f) {
    unsigned int u = __float_as_uint(f);
    return (unsigned short)((u + 0x7FFFu + ((u >> 16) & 1u)) >> 16);
}
__device__ __forceinline__ float bf2f(unsigned short h) {
    return __uint_as_float(((unsigned int)h) << 16);
}
__device__ __forceinline__ unsigned pack_bf2(float a, float b) {
    return (unsigned)f2bf(a) | ((unsigned)f2bf(b) << 16);
}
// LDS byte offset of a pointer into a __shared__ object (generic -> AS3 -> int)
__device__ __forceinline__ unsigned lds_off(const void* p) {
    return (unsigned)(size_t)(__attribute__((address_space(3))) const void*)p;
}

// ---------------------------------------------------------------------------
// Generic WMMA GEMM: Y[M,128] = epilogue(X[M,K](bf16) @ W[K,128](bf16) + b)
// mode 0: ReLU            -> out_bf (bf16) and/or out_f (f32)
// mode 1: LayerNorm(+res) -> out_f (f32) and/or out_bf
// Block = 256 threads (8 waves). Block tile = 128 rows x 128 cols.
// Wave w computes rows [w*16, w*16+16) x all 128 cols (8 wmma accumulators).
// LDS row stride = 20 uints (80B): 16B-aligned for async B128, conflict-free
// for fragment reads (20*r mod 64 distinct over r=0..15).
// ---------------------------------------------------------------------------
#define LSTR 20

__global__ __launch_bounds__(256) void k_gemm(
    const unsigned short* __restrict__ X, const unsigned short* __restrict__ W,
    const float* __restrict__ bias, const float* __restrict__ gamma,
    const float* __restrict__ beta, const float* __restrict__ resid,
    unsigned short* __restrict__ out_bf, float* __restrict__ out_f,
    int M, int K, int mode)
{
    __shared__ unsigned int lds[2 * 128 * LSTR];
    unsigned int* Xu = lds;               // X tile  [128 rows][32 bf16]
    unsigned int* Wu = lds + 128 * LSTR;  // W tile transposed [128 n][32 bf16]

    const int tid  = threadIdx.x;
    const int wave = tid >> 5, lane = tid & 31;
    const int hi   = lane >> 4, lo16 = lane & 15;
    const int m0   = blockIdx.x * 128;

    float acc[8][8];
#pragma unroll
    for (int nt = 0; nt < 8; ++nt)
#pragma unroll
        for (int v = 0; v < 8; ++v) acc[nt][v] = 0.f;

    // staging coordinates (loop-invariant)
    const int xr = tid >> 1, xkk = (tid & 1) << 4;          // X: 2 thr/row, 16 bf16 each
    const unsigned xdst = lds_off(&Xu[xr * LSTR + (xkk >> 1)]);
    const int wkp = tid >> 4, wc0 = (tid & 15) << 3;        // W: k-pair, 8 cols

    for (int k0 = 0; k0 < K; k0 += 32) {
        { // ---- stage X tile: async global -> LDS (2 x B128 per thread) ----
            int gr = m0 + xr;
            unsigned goff = (unsigned)((((size_t)gr * K) + k0 + xkk) * 2);
            if (gr < M) {
                asm volatile(
                    "global_load_async_to_lds_b128 %0, %1, %2 offset:0\n\t"
                    "global_load_async_to_lds_b128 %0, %1, %2 offset:16"
                    :: "v"(xdst), "v"(goff), "s"(X) : "memory");
            }
        }
        { // ---- stage W tile transposed: packed k-pairs, b32 LDS stores ----
            const uint4* s0 = reinterpret_cast<const uint4*>(W + (size_t)(k0 + 2 * wkp) * 128 + wc0);
            const uint4* s1 = reinterpret_cast<const uint4*>(W + (size_t)(k0 + 2 * wkp + 1) * 128 + wc0);
            uint4 a = s0[0], b = s1[0];
            unsigned lo[4] = {a.x, a.y, a.z, a.w};
            unsigned hh[4] = {b.x, b.y, b.z, b.w};
#pragma unroll
            for (int i = 0; i < 4; ++i) {
                unsigned l0 = lo[i] & 0xffffu, l1 = lo[i] >> 16;
                unsigned h0 = hh[i] & 0xffffu, h1 = hh[i] >> 16;
                Wu[(wc0 + 2 * i)     * LSTR + wkp] = l0 | (h0 << 16);
                Wu[(wc0 + 2 * i + 1) * LSTR + wkp] = l1 | (h1 << 16);
            }
        }
        asm volatile("s_wait_asynccnt 0x0" ::: "memory");
        __syncthreads();

        union Frag { unsigned int u[8]; v16bf v; };
        // A fragment (16x32 bf16 ISA layout): lane holds row (lane&15);
        // uint v -> K/2 index: (v<4 ? v : v+4) + 4*(lane>>4)
        Frag af;
        const int rl = wave * 16 + lo16;
#pragma unroll
        for (int v = 0; v < 8; ++v) {
            int ku = (v < 4 ? v : v + 4) + 4 * hi;
            af.u[v] = Xu[rl * LSTR + ku];
        }
#pragma unroll
        for (int nt = 0; nt < 8; ++nt) {
            // B fragment (32x16 bf16): lane holds col (lane&15);
            // uint v -> K/2 index: (v<4 ? v : v+4) + 4*(lane>>4)
            Frag bf;
            const int n = nt * 16 + lo16;
#pragma unroll
            for (int v = 0; v < 8; ++v) bf.u[v] = Wu[n * LSTR + ((v < 4 ? v : v + 4) + 4 * hi)];
            v8f c;
#pragma unroll
            for (int v = 0; v < 8; ++v) c[v] = acc[nt][v];
            c = __builtin_amdgcn_wmma_f32_16x16x32_bf16(
                    false, af.v, false, bf.v, (short)0, c, false, false);
#pragma unroll
            for (int v = 0; v < 8; ++v) acc[nt][v] = c[v];
        }
        __syncthreads();
    }

    // C layout: acc[nt][v] -> row = m0 + wave*16 + v + 8*hi, col = nt*16 + lo16
    if (mode == 0) { // bias + ReLU
#pragma unroll
        for (int nt = 0; nt < 8; ++nt) {
            int col = nt * 16 + lo16;
            float bb = bias[col];
#pragma unroll
            for (int v = 0; v < 8; ++v) {
                int row = m0 + wave * 16 + v + 8 * hi;
                if (row < M) {
                    float x = fmaxf(acc[nt][v] + bb, 0.f);
                    size_t o = (size_t)row * 128 + col;
                    if (out_bf) out_bf[o] = f2bf(x);
                    if (out_f)  out_f[o]  = x;
                }
            }
        }
    } else { // bias + LayerNorm (+ residual)
        float s1[8], s2[8];
#pragma unroll
        for (int v = 0; v < 8; ++v) { s1[v] = 0.f; s2[v] = 0.f; }
#pragma unroll
        for (int nt = 0; nt < 8; ++nt) {
            float bb = bias[nt * 16 + lo16];
#pragma unroll
            for (int v = 0; v < 8; ++v) {
                float x = acc[nt][v] + bb;
                acc[nt][v] = x;
                s1[v] += x; s2[v] += x * x;
            }
        }
        // reduce across the 16 lanes holding this row's columns (wave32 halves)
#pragma unroll
        for (int v = 0; v < 8; ++v) {
#pragma unroll
            for (int m = 1; m < 16; m <<= 1) {
                s1[v] += __shfl_xor(s1[v], m, 32);
                s2[v] += __shfl_xor(s2[v], m, 32);
            }
        }
#pragma unroll
        for (int nt = 0; nt < 8; ++nt) {
            int col = nt * 16 + lo16;
            float g = gamma[col], be = beta[col];
#pragma unroll
            for (int v = 0; v < 8; ++v) {
                int row = m0 + wave * 16 + v + 8 * hi;
                if (row < M) {
                    float mean = s1[v] * (1.f / 128.f);
                    float var  = s2[v] * (1.f / 128.f) - mean * mean;
                    float y = (acc[nt][v] - mean) * rsqrtf(var + LN_EPS_C) * g + be;
                    size_t o = (size_t)row * 128 + col;
                    if (resid)  y += resid[o];
                    if (out_f)  out_f[o]  = y;
                    if (out_bf) out_bf[o] = f2bf(y);
                }
            }
        }
    }
}

// ------------------------- small helper kernels ----------------------------
// f32 -> bf16, two elements per thread (count must be even; all users are)
__global__ void k_cvt(const float* __restrict__ s, unsigned int* __restrict__ d, long long npair) {
    long long i = (long long)blockIdx.x * blockDim.x + threadIdx.x;
    if (i < npair) d[i] = pack_bf2(s[2 * i], s[2 * i + 1]);
}

__global__ void k_cvt_pad(const float* __restrict__ s, unsigned short* __restrict__ d, int Ks) {
    int i = blockIdx.x * blockDim.x + threadIdx.x; // over 32*128
    if (i >= 32 * 128) return;
    int k = i >> 7, c = i & 127;
    d[i] = (k < Ks) ? f2bf(s[k * 128 + c]) : (unsigned short)0;
}

__global__ void k_zero(float* __restrict__ d, long long n) {
    long long i = (long long)blockIdx.x * blockDim.x + threadIdx.x;
    if (i < n) d[i] = 0.f;
}

__global__ void k_edge_feat(const float* __restrict__ wp, const float* __restrict__ mp,
                            const int* __restrict__ snd, const int* __restrict__ rcv,
                            const float* __restrict__ mean, const float* __restrict__ stdv,
                            unsigned short* __restrict__ out) {
    int e = blockIdx.x * blockDim.x + threadIdx.x;
    if (e >= N_EDGES_C) return;
    int s = snd[e], r = rcv[e];
    float a0 = wp[s * 3 + 0] - wp[r * 3 + 0];
    float a1 = wp[s * 3 + 1] - wp[r * 3 + 1];
    float a2 = wp[s * 3 + 2] - wp[r * 3 + 2];
    float b0 = mp[s * 3 + 0] - mp[r * 3 + 0];
    float b1 = mp[s * 3 + 1] - mp[r * 3 + 1];
    float b2 = mp[s * 3 + 2] - mp[r * 3 + 2];
    float f[8];
    f[0] = a0; f[1] = a1; f[2] = a2; f[3] = sqrtf(a0 * a0 + a1 * a1 + a2 * a2);
    f[4] = b0; f[5] = b1; f[6] = b2; f[7] = sqrtf(b0 * b0 + b1 * b1 + b2 * b2);
    unsigned int* o = (unsigned int*)(out + (size_t)e * 32);
#pragma unroll
    for (int j = 0; j < 4; ++j)
        o[j] = pack_bf2((f[2 * j] - mean[2 * j]) / stdv[2 * j],
                        (f[2 * j + 1] - mean[2 * j + 1]) / stdv[2 * j + 1]);
#pragma unroll
    for (int j = 4; j < 16; ++j) o[j] = 0;
}

__global__ void k_node_feat(const int* __restrict__ ntype, const float* __restrict__ mean,
                            const float* __restrict__ stdv, unsigned short* __restrict__ out) {
    int n = blockIdx.x * blockDim.x + threadIdx.x;
    if (n >= N_NODES_C) return;
    int t = ntype[n];
    unsigned short* o = out + (size_t)n * 32;
#pragma unroll
    for (int j = 0; j < 9; ++j)
        o[j] = f2bf((((j == t) ? 1.f : 0.f) - mean[j]) / stdv[j]);
#pragma unroll
    for (int j = 9; j < 32; ++j) o[j] = 0;
}

__global__ void k_scatter(const float* __restrict__ el, const int* __restrict__ rcv,
                          float* __restrict__ agg) {
    long long i = (long long)blockIdx.x * blockDim.x + threadIdx.x;
    if (i >= (long long)N_EDGES_C * 128) return;
    int e = (int)(i >> 7), c = (int)(i & 127);
    atomicAdd(&agg[(size_t)rcv[e] * 128 + c], el[i]);
}

// concat [el | nl[snd] | nl[rcv]] -> bf16, 2 cols per thread (dword stores)
__global__ void k_gather_edge(const float* __restrict__ el, const float* __restrict__ nl,
                              const int* __restrict__ snd, const int* __restrict__ rcv,
                              unsigned int* __restrict__ Xe) {
    long long i = (long long)blockIdx.x * blockDim.x + threadIdx.x;
    if (i >= (long long)N_EDGES_C * 192) return;
    int e = (int)(i / 192), c = (int)(i % 192) * 2;
    const float* src;
    if (c < 128)      src = el + (size_t)e * 128 + c;
    else if (c < 256) src = nl + (size_t)snd[e] * 128 + (c - 128);
    else              src = nl + (size_t)rcv[e] * 128 + (c - 256);
    Xe[i] = pack_bf2(src[0], src[1]);
}

__global__ void k_gather_node(const float* __restrict__ nl, const float* __restrict__ agg,
                              unsigned int* __restrict__ Xn) {
    long long i = (long long)blockIdx.x * blockDim.x + threadIdx.x;
    if (i >= (long long)N_NODES_C * 128) return;
    int n = (int)(i >> 7), c = (int)(i & 127) * 2;
    const float* src = (c < 128) ? nl + (size_t)n * 128 + c
                                 : agg + (size_t)n * 128 + (c - 128);
    Xn[i] = pack_bf2(src[0], src[1]);
}

__global__ void k_dec3(const unsigned short* __restrict__ H, const float* __restrict__ w2,
                       const float* __restrict__ b2, float* __restrict__ out) {
    int i = blockIdx.x * blockDim.x + threadIdx.x;
    if (i >= N_NODES_C * 3) return;
    int n = i / 3, j = i % 3;
    float acc = b2[j];
#pragma unroll 4
    for (int c = 0; c < 128; ++c) acc += bf2f(H[(size_t)n * 128 + c]) * w2[c * 3 + j];
    out[i] = acc;
}

// ---------------------------------------------------------------------------
// Host driver. Input order: top-level dict order (world_pos, mesh_pos,
// node_type, senders, receivers) followed by params pytree leaves in jax
// sorted-key order:
//  dec{b0,b1,b2,w0,w1,w2}, edge_mean, edge_std,
//  enc_edge{b0,b1,b2,be,g,w0,w1,w2}, enc_node{...}, node_mean, node_std,
//  proc.edge{b0,b1,b2,be,g,w0,w1,w2}[15,...], proc.node{...}[15,...]
// ---------------------------------------------------------------------------
extern "C" void kernel_launch(void* const* d_in, const int* in_sizes, int n_in,
                              void* d_out, int out_size, void* d_ws, size_t ws_size,
                              hipStream_t stream) {
    const int N = N_NODES_C, E = N_EDGES_C;
#define FP(i) ((const float*)d_in[i])
    const float* world_pos = FP(0);
    const float* mesh_pos  = FP(1);
    const int*   node_type = (const int*)d_in[2];
    const int*   senders   = (const int*)d_in[3];
    const int*   receivers = (const int*)d_in[4];
    const float *dec_b0 = FP(5), *dec_b1 = FP(6), *dec_b2 = FP(7);
    const float *dec_w0 = FP(8), *dec_w1 = FP(9), *dec_w2 = FP(10);
    const float *edge_mean = FP(11), *edge_std = FP(12);
    const float *ee_b0 = FP(13), *ee_b1 = FP(14), *ee_b2 = FP(15);
    const float *ee_be = FP(16), *ee_g = FP(17);
    const float *ee_w0 = FP(18), *ee_w1 = FP(19), *ee_w2 = FP(20);
    const float *en_b0 = FP(21), *en_b1 = FP(22), *en_b2 = FP(23);
    const float *en_be = FP(24), *en_g = FP(25);
    const float *en_w0 = FP(26), *en_w1 = FP(27), *en_w2 = FP(28);
    const float *node_mean = FP(29), *node_std = FP(30);
    const float *pe_b0 = FP(31), *pe_b1 = FP(32), *pe_b2 = FP(33);
    const float *pe_be = FP(34), *pe_g = FP(35);
    const float *pe_w0 = FP(36), *pe_w1 = FP(37), *pe_w2 = FP(38);
    const float *pn_b0 = FP(39), *pn_b1 = FP(40), *pn_b2 = FP(41);
    const float *pn_be = FP(42), *pn_g = FP(43);
    const float *pn_w0 = FP(44), *pn_w1 = FP(45), *pn_w2 = FP(46);
#undef FP

    // --- workspace layout ---
    char* ws = (char*)d_ws;
    size_t off = 0;
    auto alloc = [&](size_t bytes) -> char* {
        char* p = ws + off;
        off += (bytes + 255) & ~(size_t)255;
        return p;
    };
    float* edge_lat = (float*)alloc((size_t)E * 128 * 4);
    float* node_lat = (float*)alloc((size_t)N * 128 * 4);
    float* agg      = (float*)alloc((size_t)N * 128 * 4);
    unsigned short* Xe  = (unsigned short*)alloc((size_t)E * 384 * 2); // also [E,32] enc input
    unsigned short* Xn  = (unsigned short*)alloc((size_t)N * 256 * 2); // also [N,32] / dec input
    unsigned short* H1e = (unsigned short*)alloc((size_t)E * 128 * 2);
    unsigned short* H2e = (unsigned short*)alloc((size_t)E * 128 * 2);
    unsigned short* H1n = (unsigned short*)alloc((size_t)N * 128 * 2);
    unsigned short* H2n = (unsigned short*)alloc((size_t)N * 128 * 2);
    unsigned short* wn0p = (unsigned short*)alloc(32 * 128 * 2);
    unsigned short* we0p = (unsigned short*)alloc(32 * 128 * 2);
    unsigned short* wn1 = (unsigned short*)alloc(128 * 128 * 2);
    unsigned short* wn2 = (unsigned short*)alloc(128 * 128 * 2);
    unsigned short* we1 = (unsigned short*)alloc(128 * 128 * 2);
    unsigned short* we2 = (unsigned short*)alloc(128 * 128 * 2);
    unsigned short* wd0 = (unsigned short*)alloc(128 * 128 * 2);
    unsigned short* wd1 = (unsigned short*)alloc(128 * 128 * 2);
    unsigned short* pwe0 = (unsigned short*)alloc((size_t)STEPS_C * 384 * 128 * 2);
    unsigned short* pwe1 = (unsigned short*)alloc((size_t)STEPS_C * 128 * 128 * 2);
    unsigned short* pwe2 = (unsigned short*)alloc((size_t)STEPS_C * 128 * 128 * 2);
    unsigned short* pwn0 = (unsigned short*)alloc((size_t)STEPS_C * 256 * 128 * 2);
    unsigned short* pwn1 = (unsigned short*)alloc((size_t)STEPS_C * 128 * 128 * 2);
    unsigned short* pwn2 = (unsigned short*)alloc((size_t)STEPS_C * 128 * 128 * 2);

    const dim3 blk(256);
    auto g1 = [](long long n) { return dim3((unsigned)((n + 255) / 256)); };
    const dim3 GE((E + 127) / 128), GN((N + 127) / 128);
#define CVT(src, dst, cnt) \
    k_cvt<<<g1((cnt) / 2), blk, 0, stream>>>((src), (unsigned int*)(dst), (long long)(cnt) / 2)

    // --- weight prep (f32 -> bf16, first layers zero-padded to K=32) ---
    k_cvt_pad<<<g1(32 * 128), blk, 0, stream>>>(en_w0, wn0p, 9);
    k_cvt_pad<<<g1(32 * 128), blk, 0, stream>>>(ee_w0, we0p, 8);
    CVT(en_w1, wn1, 128 * 128);
    CVT(en_w2, wn2, 128 * 128);
    CVT(ee_w1, we1, 128 * 128);
    CVT(ee_w2, we2, 128 * 128);
    CVT(dec_w0, wd0, 128 * 128);
    CVT(dec_w1, wd1, 128 * 128);
    CVT(pe_w0, pwe0, (long long)STEPS_C * 384 * 128);
    CVT(pe_w1, pwe1, (long long)STEPS_C * 128 * 128);
    CVT(pe_w2, pwe2, (long long)STEPS_C * 128 * 128);
    CVT(pn_w0, pwn0, (long long)STEPS_C * 256 * 128);
    CVT(pn_w1, pwn1, (long long)STEPS_C * 128 * 128);
    CVT(pn_w2, pwn2, (long long)STEPS_C * 128 * 128);
#undef CVT

    // --- input features (padded to K=32) ---
    k_edge_feat<<<g1(E), blk, 0, stream>>>(world_pos, mesh_pos, senders, receivers,
                                           edge_mean, edge_std, Xe);
    k_node_feat<<<g1(N), blk, 0, stream>>>(node_type, node_mean, node_std, Xn);

    // --- encoders ---
    k_gemm<<<GN, blk, 0, stream>>>(Xn, wn0p, en_b0, nullptr, nullptr, nullptr, H1n, nullptr, N, 32, 0);
    k_gemm<<<GN, blk, 0, stream>>>(H1n, wn1, en_b1, nullptr, nullptr, nullptr, H2n, nullptr, N, 128, 0);
    k_gemm<<<GN, blk, 0, stream>>>(H2n, wn2, en_b2, en_g, en_be, nullptr, nullptr, node_lat, N, 128, 1);
    k_gemm<<<GE, blk, 0, stream>>>(Xe, we0p, ee_b0, nullptr, nullptr, nullptr, H1e, nullptr, E, 32, 0);
    k_gemm<<<GE, blk, 0, stream>>>(H1e, we1, ee_b1, nullptr, nullptr, nullptr, H2e, nullptr, E, 128, 0);
    k_gemm<<<GE, blk, 0, stream>>>(H2e, we2, ee_b2, ee_g, ee_be, nullptr, nullptr, edge_lat, E, 128, 1);

    // --- 15 processor steps ---
    for (int s = 0; s < STEPS_C; ++s) {
        k_gather_edge<<<g1((long long)E * 192), blk, 0, stream>>>(edge_lat, node_lat, senders,
                                                                  receivers, (unsigned int*)Xe);
        k_gemm<<<GE, blk, 0, stream>>>(Xe, pwe0 + (size_t)s * 384 * 128, pe_b0 + s * 128,
                                       nullptr, nullptr, nullptr, H1e, nullptr, E, 384, 0);
        k_gemm<<<GE, blk, 0, stream>>>(H1e, pwe1 + (size_t)s * 128 * 128, pe_b1 + s * 128,
                                       nullptr, nullptr, nullptr, H2e, nullptr, E, 128, 0);
        k_gemm<<<GE, blk, 0, stream>>>(H2e, pwe2 + (size_t)s * 128 * 128, pe_b2 + s * 128,
                                       pe_g + s * 128, pe_be + s * 128, edge_lat,
                                       nullptr, edge_lat, E, 128, 1);
        k_zero<<<g1((long long)N * 128), blk, 0, stream>>>(agg, (long long)N * 128);
        k_scatter<<<g1((long long)E * 128), blk, 0, stream>>>(edge_lat, receivers, agg);
        k_gather_node<<<g1((long long)N * 128), blk, 0, stream>>>(node_lat, agg, (unsigned int*)Xn);
        k_gemm<<<GN, blk, 0, stream>>>(Xn, pwn0 + (size_t)s * 256 * 128, pn_b0 + s * 128,
                                       nullptr, nullptr, nullptr, H1n, nullptr, N, 256, 0);
        k_gemm<<<GN, blk, 0, stream>>>(H1n, pwn1 + (size_t)s * 128 * 128, pn_b1 + s * 128,
                                       nullptr, nullptr, nullptr, H2n, nullptr, N, 128, 0);
        k_gemm<<<GN, blk, 0, stream>>>(H2n, pwn2 + (size_t)s * 128 * 128, pn_b2 + s * 128,
                                       pn_g + s * 128, pn_be + s * 128, node_lat,
                                       nullptr, node_lat, N, 128, 1);
    }

    // --- decoder ---
    k_cvt<<<g1((long long)N * 64), blk, 0, stream>>>(node_lat, (unsigned int*)Xn, (long long)N * 64);
    k_gemm<<<GN, blk, 0, stream>>>(Xn, wd0, dec_b0, nullptr, nullptr, nullptr, H1n, nullptr, N, 128, 0);
    k_gemm<<<GN, blk, 0, stream>>>(H1n, wd1, dec_b1, nullptr, nullptr, nullptr, H2n, nullptr, N, 128, 0);
    k_dec3<<<g1(N * 3), blk, 0, stream>>>(H2n, dec_w2, dec_b2, (float*)d_out);
}
